// MultiHeadSelfAttention_65884798321137
// MI455X (gfx1250) — compile-verified, matrix operand order checked by reading, not run on hardware
//
#include <hip/hip_runtime.h>

// MHA on MI455X (gfx1250): compute-bound (~69 GFLOP vs ~2us of HBM traffic).
// All GEMM-shaped work -> v_wmma_f32_16x16x32_f16 (f16 operands, f32 acc).
// Pure-f16 tile staging (K tiles, ctx tiles) -> Tensor Data Mover DMA with
// TENSORcnt sync; out_proj double-buffers the TDM against WMMA compute.
// wave32, 8 waves / 256-thread block.

#define S_LEN 2048
#define B_SZ  2
#define DM    1024
#define HH    16
#define DK    64
#define NTOK  (B_SZ * S_LEN)   // 4096

typedef __attribute__((ext_vector_type(16))) _Float16 v16h;
typedef __attribute__((ext_vector_type(8)))  _Float16 v8h;
typedef __attribute__((ext_vector_type(8)))  float    v8f;
typedef __attribute__((ext_vector_type(4)))  float    v4f;

#if __has_builtin(__builtin_amdgcn_tensor_load_to_lds) && \
    __has_builtin(__builtin_amdgcn_s_wait_tensorcnt)
#define USE_TDM 1
#else
#define USE_TDM 0
#endif

#if USE_TDM
typedef unsigned int v4u __attribute__((ext_vector_type(4)));
typedef int          v4i __attribute__((ext_vector_type(4)));
typedef int          v8i __attribute__((ext_vector_type(8)));

// Issue a TDM load of a 2D f16 tile (tile_h rows x tile_w elems, row stride
// row_stride elems) from global into LDS (rows packed contiguously).
// D# layout per CDNA5 ISA ch.8: group0 = count/lds_addr/global_addr/type=2,
// group1 = data_size(2B), tensor_dim0/1, tile_dim0/1, tensor_dim0_stride.
// This toolchain exposes the 6-arg builtin (extra int32x8 group + cpol).
__device__ __forceinline__ void tdm_load_tile_f16(
    const _Float16* gsrc, const _Float16* lds_dst,
    unsigned tile_w, unsigned tile_h,
    unsigned tensor_w, unsigned tensor_h, unsigned row_stride) {
  unsigned long long ga = (unsigned long long)(uintptr_t)gsrc;
  // Flat shared pointers carry the LDS byte offset in their low 32 bits
  unsigned lds = (unsigned)(uintptr_t)lds_dst;
  v4u g0;
  g0[0] = 1u;                                                  // count=1, user
  g0[1] = lds;                                                 // lds_addr
  g0[2] = (unsigned)ga;                                        // gaddr[31:0]
  g0[3] = (unsigned)((ga >> 32) & 0x1FFFFFFull) | (2u << 30);  // gaddr hi,type=2
  v8i g1;
  g1[0] = (int)(1u << 16);                                     // data_size=2B
  g1[1] = (int)((tensor_w & 0xFFFFu) << 16);                   // tdim0 lo16
  g1[2] = (int)(((tensor_w >> 16) & 0xFFFFu) |
                ((tensor_h & 0xFFFFu) << 16));                 // tdim0 hi|tdim1 lo
  g1[3] = (int)(((tensor_h >> 16) & 0xFFFFu) |
                ((tile_w & 0xFFFFu) << 16));                   // tdim1 hi|tile0
  g1[4] = (int)(tile_h & 0xFFFFu);                             // tile1 (tile2=0)
  g1[5] = (int)row_stride;                                     // dim0 stride lo32
  g1[6] = 0;                                                   // stride hi / d1s
  g1[7] = 0;
  v4i g2 = {0, 0, 0, 0};                                       // 2D: dims 2+ unused
  v4i g3 = {0, 0, 0, 0};
  v8i g4 = {0, 0, 0, 0, 0, 0, 0, 0};                           // unused extra group
  __builtin_amdgcn_tensor_load_to_lds(g0, g1, g2, g3, g4, 0);
}
#endif

// A/B fragment loader for V_WMMA_F32_16X16X32_F16.
// ISA layout (16-bit A 16x32): lane = (half<<4)|m ; elements 0..7 cover
// K = half*8 + 0..7, elements 8..15 cover K = 16 + half*8 + 0..7.
// Both runs are 16B-contiguous -> two b128 loads.
__device__ __forceinline__ v16h load_frag(const _Float16* base, int stride,
                                          int row, int half) {
  const _Float16* p = base + row * stride + half * 8;
  v8h lo = *(const v8h*)(p);
  v8h hi = *(const v8h*)(p + 16);
  v16h r;
#pragma unroll
  for (int i = 0; i < 8; ++i) { r[i] = lo[i]; r[8 + i] = hi[i]; }
  return r;
}

__device__ __forceinline__ v8f wmma_f16(v16h a, v16h b, v8f c) {
  return __builtin_amdgcn_wmma_f32_16x16x32_f16(false, a, false, b,
                                                (short)0, c, false, false);
}

// ---------------------------------------------------------------------------
// Kernel 1: Q/K/V = relu(x @ W[h] + b[h]), stored f16 as [token][h*64+d].
// Per block: 128 tokens x 64 cols (one head, one of Q/K/V via blockIdx.z).
// f32->f16 converting stages (no TDM: DMA cannot convert).
// ---------------------------------------------------------------------------
__global__ __launch_bounds__(256) void qkv_proj(
    const float* __restrict__ x,
    const float* __restrict__ Wq, const float* __restrict__ bq,
    const float* __restrict__ Wk, const float* __restrict__ bk,
    const float* __restrict__ Wv, const float* __restrict__ bv,
    _Float16* __restrict__ Qb, _Float16* __restrict__ Kb,
    _Float16* __restrict__ Vb) {
  __shared__ __align__(16) _Float16 As[128 * 32];  // x tile, f16
  __shared__ __align__(16) _Float16 Bs[64 * 32];   // W tile transposed [n][k]

  const int h    = blockIdx.y;
  const int sel  = blockIdx.z;
  const int tok0 = blockIdx.x * 128;

  const float* W    = (sel == 0) ? Wq : (sel == 1) ? Wk : Wv;
  const float* bias = (sel == 0) ? bq : (sel == 1) ? bk : bv;
  _Float16*    Out  = (sel == 0) ? Qb : (sel == 1) ? Kb : Vb;
  const float* Wh   = W + (size_t)h * DM * DK;

  const int tid  = threadIdx.x;
  const int wv   = tid >> 5;
  const int lane = tid & 31;
  const int lm   = lane & 15;
  const int half = lane >> 4;
  const int m0   = wv * 16;

  v8f acc[4];
#pragma unroll
  for (int t = 0; t < 4; ++t)
#pragma unroll
    for (int i = 0; i < 8; ++i) acc[t][i] = 0.0f;

  for (int k0 = 0; k0 < DM; k0 += 32) {
    {  // Stage A: x tile f32 -> f16
      const int r = tid >> 3;
      const int c = (tid & 7) * 4;
#pragma unroll
      for (int rr = 0; rr < 4; ++rr) {
        const int row = rr * 32 + r;
        v4f xv = *(const v4f*)(x + (size_t)(tok0 + row) * DM + k0 + c);
#pragma unroll
        for (int j = 0; j < 4; ++j) As[row * 32 + c + j] = (_Float16)xv[j];
      }
    }
    {  // Stage B transposed: Wh[k0+row][c] -> Bs[c][row]
      const int r = tid >> 4;
      const int c = (tid & 15) * 4;
#pragma unroll
      for (int rr = 0; rr < 2; ++rr) {
        const int row = rr * 16 + r;
        v4f wv4 = *(const v4f*)(Wh + (size_t)(k0 + row) * DK + c);
#pragma unroll
        for (int j = 0; j < 4; ++j) Bs[(c + j) * 32 + row] = (_Float16)wv4[j];
      }
    }
    __syncthreads();
    v16h a = load_frag(As, 32, m0 + lm, half);
#pragma unroll
    for (int t = 0; t < 4; ++t) {
      v16h b = load_frag(Bs, 32, t * 16 + lm, half);
      acc[t] = wmma_f16(a, b, acc[t]);
    }
    __syncthreads();
  }

#pragma unroll
  for (int t = 0; t < 4; ++t) {
    const int n = t * 16 + lm;
    const float bv_ = bias[h * DK + n];
#pragma unroll
    for (int v = 0; v < 8; ++v) {
      const int row = tok0 + m0 + v + 8 * half;
      float val = fmaxf(acc[t][v] + bv_, 0.0f);
      Out[(size_t)row * DM + h * DK + n] = (_Float16)val;
    }
  }
}

// ---------------------------------------------------------------------------
// Kernel 2: flash attention. Block = (b,h) x 128 queries; wave = 16 queries.
// Online softmax; QK^T and P.V via WMMA. 32 keys per iteration.
// K tile staged by TDM (pure f16 tile copy); V transposed manually.
// ---------------------------------------------------------------------------
__global__ __launch_bounds__(256) void attn(
    const _Float16* __restrict__ Qb, const _Float16* __restrict__ Kb,
    const _Float16* __restrict__ Vb, _Float16* __restrict__ Ctx) {
  __shared__ __align__(16) _Float16 Ks[32 * 64];      // K tile [key][d]
  __shared__ __align__(16) _Float16 Vt[64 * 32];      // V tile transposed [d][key]
  __shared__ __align__(16) _Float16 Ps[8 * 16 * 32];  // per-wave P scratch

  const int q0   = blockIdx.x * 128;
  const int h    = blockIdx.y;
  const int b    = blockIdx.z;
  const int tid  = threadIdx.x;
  const int wv   = tid >> 5;
  const int lane = tid & 31;
  const int lm   = lane & 15;
  const int half = lane >> 4;
  const size_t tokBase = (size_t)b * S_LEN;

  v16h qf[2];
  {
    const _Float16* qrow = Qb + (tokBase + q0 + wv * 16 + lm) * DM + h * DK;
#pragma unroll
    for (int kc = 0; kc < 2; ++kc) qf[kc] = load_frag(qrow + kc * 32, 0, 0, half);
  }

  float m_run[8], l_run[8];
  v8f acc[4];
#pragma unroll
  for (int v = 0; v < 8; ++v) { m_run[v] = -1.0e30f; l_run[v] = 0.0f; }
#pragma unroll
  for (int t = 0; t < 4; ++t)
#pragma unroll
    for (int i = 0; i < 8; ++i) acc[t][i] = 0.0f;

  _Float16* PsW = Ps + wv * 512;

  for (int kb = 0; kb < S_LEN / 32; ++kb) {
    const _Float16* ksrc = Kb + (tokBase + kb * 32) * DM + h * DK;
#if USE_TDM
    if (wv == 0) {
      // 32 rows x 64 f16, row stride 1024 elems -> Ks rows packed.
      tdm_load_tile_f16(ksrc, Ks, /*tile_w=*/64, /*tile_h=*/32,
                        /*tensor_w=*/64, /*tensor_h=*/32, /*stride=*/DM);
    }
#else
    {
      const int r  = tid >> 3;
      const int c8 = (tid & 7) * 8;
      *(v8h*)(Ks + r * 64 + c8) = *(const v8h*)(ksrc + (size_t)r * DM + c8);
    }
#endif
    {  // Stage V transposed (overlaps the TDM)
      const int r  = tid >> 3;
      const int c8 = (tid & 7) * 8;
      v8h vvv = *(const v8h*)(Vb + (tokBase + kb * 32 + r) * DM + h * DK + c8);
#pragma unroll
      for (int j = 0; j < 8; ++j) Vt[(c8 + j) * 32 + r] = vvv[j];
    }
#if USE_TDM
    if (wv == 0) __builtin_amdgcn_s_wait_tensorcnt(0);
#endif
    __syncthreads();

    // S = Q . K^T
    v8f s0, s1;
#pragma unroll
    for (int i = 0; i < 8; ++i) { s0[i] = 0.0f; s1[i] = 0.0f; }
#pragma unroll
    for (int kc = 0; kc < 2; ++kc) {
      v16h b0 = load_frag(Ks + kc * 32, 64, lm, half);
      v16h b1 = load_frag(Ks + kc * 32, 64, 16 + lm, half);
      s0 = wmma_f16(qf[kc], b0, s0);
      s1 = wmma_f16(qf[kc], b1, s1);
    }

    // Online softmax (row m = v + 8*half lives across a 16-lane half-group)
#pragma unroll
    for (int v = 0; v < 8; ++v) {
      float a0 = s0[v] * 0.125f;  // 1/sqrt(64)
      float a1 = s1[v] * 0.125f;
      float mx = fmaxf(a0, a1);
#pragma unroll
      for (int msk = 8; msk >= 1; msk >>= 1)
        mx = fmaxf(mx, __shfl_xor(mx, msk, 32));
      const float mnew  = fmaxf(m_run[v], mx);
      const float alpha = __expf(m_run[v] - mnew);
      const float p0 = __expf(a0 - mnew);
      const float p1 = __expf(a1 - mnew);
      float rs = p0 + p1;
#pragma unroll
      for (int msk = 8; msk >= 1; msk >>= 1) rs += __shfl_xor(rs, msk, 32);
      l_run[v] = l_run[v] * alpha + rs;
      m_run[v] = mnew;
#pragma unroll
      for (int t = 0; t < 4; ++t) acc[t][v] *= alpha;
      s0[v] = p0;
      s1[v] = p1;
    }

    // C-layout -> A-fragment layout via per-wave LDS scratch
#pragma unroll
    for (int v = 0; v < 8; ++v) {
      const int row = v + 8 * half;
      PsW[row * 32 + lm]      = (_Float16)s0[v];
      PsW[row * 32 + 16 + lm] = (_Float16)s1[v];
    }
    __syncthreads();

    v16h pf = load_frag(PsW, 32, lm, half);
#pragma unroll
    for (int t = 0; t < 4; ++t) {
      v16h vf = load_frag(Vt, 32, t * 16 + lm, half);
      acc[t] = wmma_f16(pf, vf, acc[t]);
    }
    __syncthreads();
  }

#pragma unroll
  for (int v = 0; v < 8; ++v) {
    const float inv = 1.0f / l_run[v];
    const int row = q0 + wv * 16 + v + 8 * half;
#pragma unroll
    for (int t = 0; t < 4; ++t) {
      Ctx[(tokBase + row) * DM + h * DK + t * 16 + lm] =
          (_Float16)(acc[t][v] * inv);
    }
  }
}

// ---------------------------------------------------------------------------
// Kernel 3: out = relu(ctx @ Wo + bo), M=4096 K=1024 N=1024, f32 output.
// ctx A-tiles DMA'd by TDM, double-buffered against WMMA compute.
// ---------------------------------------------------------------------------
__global__ __launch_bounds__(256) void out_proj(
    const _Float16* __restrict__ Ctx, const float* __restrict__ Wo,
    const float* __restrict__ bo, float* __restrict__ Out) {
  __shared__ __align__(16) _Float16 As[2][128 * 32];
  __shared__ __align__(16) _Float16 Bs[64 * 32];

  const int tok0 = blockIdx.x * 128;
  const int n0   = blockIdx.y * 64;
  const int tid  = threadIdx.x;
  const int wv   = tid >> 5;
  const int lane = tid & 31;
  const int lm   = lane & 15;
  const int half = lane >> 4;
  const int m0   = wv * 16;

  v8f acc[4];
#pragma unroll
  for (int t = 0; t < 4; ++t)
#pragma unroll
    for (int i = 0; i < 8; ++i) acc[t][i] = 0.0f;

#if USE_TDM
  if (wv == 0) {  // prime the pipeline: tile k0=0 -> As[0]
    tdm_load_tile_f16(Ctx + (size_t)tok0 * DM, As[0], 32, 128, 32, 128, DM);
  }
#endif

  for (int k0 = 0; k0 < DM; k0 += 32) {
    const int cur = (k0 >> 5) & 1;
#if USE_TDM
    if (wv == 0 && k0 + 32 < DM) {  // prefetch next tile into the other buffer
      tdm_load_tile_f16(Ctx + (size_t)tok0 * DM + k0 + 32, As[cur ^ 1],
                        32, 128, 32, 128, DM);
    }
#else
    {
      const int r   = tid >> 1;
      const int c16 = (tid & 1) * 16;
      const _Float16* src = Ctx + (size_t)(tok0 + r) * DM + k0 + c16;
      *(v8h*)(As[cur] + r * 32 + c16)     = *(const v8h*)(src);
      *(v8h*)(As[cur] + r * 32 + c16 + 8) = *(const v8h*)(src + 8);
    }
#endif
    {  // Stage B transposed: Wo[k0+row][n0+c] -> Bs[c][row] (overlaps TDM)
      const int r = tid >> 4;
      const int c = (tid & 15) * 4;
#pragma unroll
      for (int rr = 0; rr < 2; ++rr) {
        const int row = rr * 16 + r;
        v4f w = *(const v4f*)(Wo + (size_t)(k0 + row) * DM + n0 + c);
#pragma unroll
        for (int j = 0; j < 4; ++j) Bs[(c + j) * 32 + row] = (_Float16)w[j];
      }
    }
#if USE_TDM
    // TDM completes in order per wave: allow the prefetch to stay in flight.
    // s_wait_tensorcnt requires an immediate operand.
    if (wv == 0) {
      if (k0 + 32 < DM) {
        __builtin_amdgcn_s_wait_tensorcnt(1);
      } else {
        __builtin_amdgcn_s_wait_tensorcnt(0);
      }
    }
#endif
    __syncthreads();
    v16h a = load_frag(As[cur], 32, m0 + lm, half);
#pragma unroll
    for (int t = 0; t < 4; ++t) {
      v16h b = load_frag(Bs, 32, t * 16 + lm, half);
      acc[t] = wmma_f16(a, b, acc[t]);
    }
    __syncthreads();
  }

#pragma unroll
  for (int t = 0; t < 4; ++t) {
    const int n = n0 + t * 16 + lm;
    const float bb = bo[n];
#pragma unroll
    for (int v = 0; v < 8; ++v) {
      const int row = tok0 + m0 + v + 8 * half;
      Out[(size_t)row * DM + n] = fmaxf(acc[t][v] + bb, 0.0f);
    }
  }
}

// ---------------------------------------------------------------------------
extern "C" void kernel_launch(void* const* d_in, const int* in_sizes, int n_in,
                              void* d_out, int out_size, void* d_ws,
                              size_t ws_size, hipStream_t stream) {
  const float* x  = (const float*)d_in[0];
  const float* Wq = (const float*)d_in[1];
  const float* bq = (const float*)d_in[2];
  const float* Wk = (const float*)d_in[3];
  const float* bk = (const float*)d_in[4];
  const float* Wv = (const float*)d_in[5];
  const float* bv = (const float*)d_in[6];
  const float* Wo = (const float*)d_in[7];
  const float* bo = (const float*)d_in[8];
  float* out = (float*)d_out;

  _Float16* qb = (_Float16*)d_ws;
  _Float16* kb = qb + (size_t)NTOK * DM;
  _Float16* vb = kb + (size_t)NTOK * DM;
  _Float16* cb = vb + (size_t)NTOK * DM;

  qkv_proj<<<dim3(NTOK / 128, HH, 3), 256, 0, stream>>>(
      x, Wq, bq, Wk, bk, Wv, bv, qb, kb, vb);
  attn<<<dim3(S_LEN / 128, HH, B_SZ), 256, 0, stream>>>(qb, kb, vb, cb);
  out_proj<<<dim3(NTOK / 128, DM / 64), 256, 0, stream>>>(cb, Wo, bo, out);
}
